// Encoder_16243566313796
// MI455X (gfx1250) — matrix operand order repaired and checked by reading, not compiled, not run on hardware
//
#include <hip/hip_runtime.h>

// ---------------------------------------------------------------------------
// GRU encoder for MI455X (gfx1250): fused embedding-gather + input projection
// + recurrence. Matmuls via v_wmma_f32_16x16x32_bf16 (wave32); tile staging
// via Tensor Data Mover (tensor_load_to_lds, incl. gather mode) when the
// toolchain exposes the builtin, with a manual LDS-copy fallback.
// ---------------------------------------------------------------------------

#define Bsz 64
#define Ssz 2048
#define Dsz 256
#define Hsz 512
#define G3  1536   // 3*H
#define Vsz 50000

typedef __attribute__((ext_vector_type(16))) __bf16 v16bf;
typedef __attribute__((ext_vector_type(8)))  __bf16 bf16x8;
typedef __attribute__((ext_vector_type(8)))  float  v8f;
typedef __attribute__((ext_vector_type(4)))  unsigned int u32x4;
typedef __attribute__((ext_vector_type(8)))  int i32x8;
typedef __attribute__((ext_vector_type(4)))  int i32x4;

#if defined(__has_builtin)
#if __has_builtin(__builtin_amdgcn_tensor_load_to_lds)
#define HAVE_TDM 1
#endif
#if __has_builtin(__builtin_amdgcn_s_wait_tensorcnt)
#define HAVE_TENSORCNT 1
#endif
#endif

static __device__ inline __bf16 f2bf(float f) {
    unsigned int u = __builtin_bit_cast(unsigned int, f);
    unsigned int r = (u + 0x7FFFu + ((u >> 16) & 1u)) >> 16;   // round-nearest-even
    return __builtin_bit_cast(__bf16, (unsigned short)r);
}

static __device__ inline float sigmoid_f(float x) {
    return 1.0f / (1.0f + __expf(-x));
}
static __device__ inline float tanh_f(float x) {
    x = fminf(fmaxf(x, -15.0f), 15.0f);
    float e = __expf(2.0f * x);
    return (e - 1.0f) / (e + 1.0f);
}

static __device__ inline void wait_tensorcnt0() {
#if defined(HAVE_TENSORCNT)
    __builtin_amdgcn_s_wait_tensorcnt(0);
#else
    asm volatile("s_wait_tensorcnt 0x0" ::: "memory");
#endif
}

#if defined(HAVE_TDM)
// Build a D# per CDNA5 ISA ch.8 and issue TENSOR_LOAD_TO_LDS.
//  - 2D tile load when gather==false (g2/g3 zero, dims in group1/2)
//  - gather mode when gather==true (g2/g3 carry 16x 16-bit row indices)
// data_size fixed at 2 bytes (bf16).
static __device__ inline void tdm_load(unsigned int lds_addr, const void* gaddr,
                                       unsigned int tile_d0, unsigned int tile_d1,
                                       unsigned int tensor_d0, unsigned int tensor_d1,
                                       unsigned int stride0, bool gather,
                                       i32x4 g2, i32x4 g3) {
    unsigned long long ga = (unsigned long long)(uintptr_t)gaddr;
    u32x4 g0;
    g0[0] = 1u | (gather ? (1u << 31) : 0u);          // count=1 [+gather_mode, 16b idx]
    g0[1] = lds_addr;                                  // D#.lds_addr
    g0[2] = (unsigned int)(ga & 0xFFFFFFFFu);          // global_addr[31:0]
    g0[3] = (unsigned int)((ga >> 32) & 0x01FFFFFFu)   // global_addr[56:32]
            | (2u << 30);                              // type = 2 ("image")
    i32x8 g1;
    g1[0] = 0x00010000;                                // wg_mask=0, data_size=1 (2B)
    g1[1] = (int)((tensor_d0 & 0xFFFFu) << 16);        // [63:48] tensor_dim0 lo16
    g1[2] = (int)((tensor_d0 >> 16) | ((tensor_d1 & 0xFFFFu) << 16)); // dim0 hi / dim1 lo
    g1[3] = (int)((tensor_d1 >> 16) | (tile_d0 << 16));               // dim1 hi / tile_dim0
    g1[4] = (int)(tile_d1 & 0xFFFFu);                  // tile_dim1 (tile_dim2 = 0)
    g1[5] = (int)stride0;                              // tensor_dim0_stride[31:0]
    g1[6] = 0;
    g1[7] = 0;
#if __clang_major__ >= 23
    i32x8 gpad = {0, 0, 0, 0, 0, 0, 0, 0};
    __builtin_amdgcn_tensor_load_to_lds(g0, g1, g2, g3, gpad, 0);
#else
    __builtin_amdgcn_tensor_load_to_lds(g0, g1, g2, g3, 0);
#endif
}
#endif // HAVE_TDM

// K-chunked bf16 WMMA accumulation (A from LDS, B from global bf16 weights).
static __device__ inline v8f wmma_accum(v8f acc, const __bf16* arow,
                                        const __bf16* bcol, int nchunks) {
#pragma unroll 4
    for (int kc = 0; kc < nchunks; ++kc) {
        union { v16bf v; bf16x8 h[2]; } a, b;
        a.h[0] = *(const bf16x8*)(arow + kc * 32);        // K 0..7   (+half*16)
        a.h[1] = *(const bf16x8*)(arow + kc * 32 + 8);    // K 8..15  (+half*16)
        b.h[0] = *(const bf16x8*)(bcol + kc * 32);        // K 0..7   (+half*16)
        b.h[1] = *(const bf16x8*)(bcol + kc * 32 + 8);    // K 8..15  (+half*16)
        acc = __builtin_amdgcn_wmma_f32_16x16x32_bf16(
            false, a.v, false, b.v, (short)0, acc, false, false);
    }
    return acc;
}

// ---------------------------------------------------------------------------
// Prep (one pass, grid-stride):
//   WiT [3H][D] bf16  <- transpose+convert Wi
//   WhT [3H][H] bf16  <- transpose+convert Wh
//   h f32 double-buffer = 0 ; h bf16 double-buffer = 0
//   embbf [V][D] bf16 <- convert emb
// ---------------------------------------------------------------------------
#define NWi  (G3 * Dsz)            // 393216
#define NWh  (G3 * Hsz)            // 786432
#define NHf  (2 * Bsz * Hsz)       // 65536
#define NHb  (2 * Bsz * Hsz)       // 65536
#define NEmb (Vsz * Dsz)           // 12800000
#define PREP_TOTAL (NWi + NWh + NHf + NHb + NEmb)

__global__ void prep_kernel(const float* __restrict__ Wi,
                            const float* __restrict__ Wh,
                            const float* __restrict__ emb,
                            unsigned short* __restrict__ WiT_raw,
                            unsigned short* __restrict__ WhT_raw,
                            float* __restrict__ hbufs,
                            unsigned short* __restrict__ hbf,
                            unsigned short* __restrict__ embbf_raw) {
    __bf16* WiT = (__bf16*)WiT_raw;
    __bf16* WhT = (__bf16*)WhT_raw;
    __bf16* embbf = (__bf16*)embbf_raw;
    for (long long idx = blockIdx.x * (long long)blockDim.x + threadIdx.x;
         idx < PREP_TOTAL; idx += (long long)gridDim.x * blockDim.x) {
        if (idx < NWi) {
            int n = (int)idx >> 8, k = (int)idx & 255;        // WiT[n][k] = Wi[k][n]
            WiT[idx] = f2bf(Wi[k * G3 + n]);
        } else if (idx < NWi + NWh) {
            int t = (int)(idx - NWi);
            int n = t >> 9, k = t & 511;                      // WhT[n][k] = Wh[k][n]
            WhT[t] = f2bf(Wh[k * G3 + n]);
        } else if (idx < NWi + NWh + NHf) {
            hbufs[idx - NWi - NWh] = 0.0f;
        } else if (idx < NWi + NWh + NHf + NHb) {
            hbf[idx - NWi - NWh - NHf] = 0;
        } else {
            long long e = idx - (NWi + NWh + NHf + NHb);
            embbf[e] = f2bf(emb[e]);
        }
    }
}

// ---------------------------------------------------------------------------
// One GRU timestep. Grid: (32 j-tiles, 4 m-tiles), 128 threads = 4 waves.
//   wave 0: r gate,  acc = (x@Wi + h@Wh)[:, j0:j0+16]
//   wave 1: z gate,  acc = (x@Wi + h@Wh)[:, H+j0 : ...]
//   wave 2: n gate hidden part, acc = (h@Wh)[:, 2H+j0 : ...]
//   wave 3: n gate input part,  acc = (x@Wi)[:, 2H+j0 : ...]
// h/x tiles staged into LDS by TDM (2D tile + gather-mode) when available.
// ---------------------------------------------------------------------------
__global__ __launch_bounds__(128)
void gru_step_kernel(const long long* __restrict__ tokens,
                     const unsigned short* __restrict__ embbf_raw,
                     const unsigned short* __restrict__ WiT_raw,
                     const unsigned short* __restrict__ WhT_raw,
                     const float* __restrict__ bi,
                     const float* __restrict__ bhn,
                     const float* __restrict__ hin,
                     const unsigned short* __restrict__ hbf_in_raw,
                     float* __restrict__ hout,
                     unsigned short* __restrict__ hbf_out_raw,
                     float* __restrict__ out,
                     int s) {
    const __bf16* embbf = (const __bf16*)embbf_raw;
    const __bf16* WiT = (const __bf16*)WiT_raw;
    const __bf16* WhT = (const __bf16*)WhT_raw;
    const __bf16* hbf_in = (const __bf16*)hbf_in_raw;
    __bf16* hbf_out = (__bf16*)hbf_out_raw;

    __shared__ __bf16 sH[16 * Hsz];        // h tile      (16 KB)
    __shared__ __bf16 sX[16 * Dsz];        // emb tile    ( 8 KB)
    __shared__ float  sAcc[4][16][16];     // gate accs   ( 4 KB)
    __shared__ long long stok[16];

    const int tid  = threadIdx.x;
    const int lane = tid & 31;
    const int wv   = tid >> 5;
    const int j0   = blockIdx.x * 16;      // hidden-unit tile
    const int m0   = blockIdx.y * 16;      // batch tile

    if (tid < 16)
        stok[tid] = tokens[(long long)(m0 + tid) * Ssz + s];
    __syncthreads();

#if defined(HAVE_TDM)
    if (wv == 0) {
        // h tile: plain 2D 16x512 bf16 tile of the 64x512 h tensor
        i32x4 gz = {0, 0, 0, 0};
        tdm_load((unsigned int)(uintptr_t)&sH[0], hbf_in + (size_t)m0 * Hsz,
                 /*tile*/ Hsz, 16, /*tensor*/ Hsz, Bsz, /*stride0*/ Hsz,
                 false, gz, gz);
        // emb tile: TDM gather mode, 16 x 16-bit row indices (V < 65536)
        i32x4 g2, g3;
#pragma unroll
        for (int i = 0; i < 4; ++i) {
            g2[i] = (int)(((unsigned int)stok[2 * i] & 0xFFFFu) |
                          (((unsigned int)stok[2 * i + 1] & 0xFFFFu) << 16));
            g3[i] = (int)(((unsigned int)stok[8 + 2 * i] & 0xFFFFu) |
                          (((unsigned int)stok[9 + 2 * i] & 0xFFFFu) << 16));
        }
        tdm_load((unsigned int)(uintptr_t)&sX[0], embbf,
                 /*tile*/ Dsz, 16, /*tensor*/ Dsz, Vsz, /*stride0*/ Dsz,
                 true, g2, g3);
        wait_tensorcnt0();
    }
#else
    // Fallback: manual 16-byte LDS staging (pure byte movement, bf16 sources)
#pragma unroll
    for (int i = 0; i < 8; ++i) {          // 16x512 bf16 = 1024 x 16B chunks
        int c = i * 128 + tid;
        int r = c >> 6, k8 = c & 63;
        *(uint4*)&sH[r * Hsz + k8 * 8] =
            *(const uint4*)&hbf_in[(size_t)(m0 + r) * Hsz + k8 * 8];
    }
#pragma unroll
    for (int i = 0; i < 4; ++i) {          // 16x256 bf16 = 512 x 16B chunks
        int c = i * 128 + tid;
        int r = c >> 5, k8 = c & 31;
        *(uint4*)&sX[r * Dsz + k8 * 8] =
            *(const uint4*)&embbf[(size_t)stok[r] * Dsz + k8 * 8];
    }
#endif
    __syncthreads();

    const int hl  = lane >> 4;   // 0 = lanes 0-15, 1 = lanes 16-31
    const int l16 = lane & 15;

    // A fragment base pointers (contiguous k: lane half h holds k = 16h..16h+15 of each chunk)
    const __bf16* aH = &sH[l16 * Hsz + hl * 16];
    const __bf16* aX = &sX[l16 * Dsz + hl * 16];

    if (wv < 3) {
        v8f acc = {};
        const __bf16* bH = WhT + (size_t)(wv * Hsz + j0 + l16) * Hsz + hl * 16;
        acc = wmma_accum(acc, aH, bH, Hsz / 32);           // h @ Wh, 16 chunks
        if (wv < 2) {                                      // fuse x@Wi for r,z
            const __bf16* bX = WiT + (size_t)(wv * Hsz + j0 + l16) * Dsz + hl * 16;
            acc = wmma_accum(acc, aX, bX, Dsz / 32);       // 8 chunks
        }
#pragma unroll
        for (int v = 0; v < 8; ++v)
            sAcc[wv][v + hl * 8][l16] = acc[v];            // C: M = v+half*8, N = l16
    } else {
        v8f acc = {};
        const __bf16* bX = WiT + (size_t)(2 * Hsz + j0 + l16) * Dsz + hl * 16;
        acc = wmma_accum(acc, aX, bX, Dsz / 32);           // x @ Wi for n gate
#pragma unroll
        for (int v = 0; v < 8; ++v)
            sAcc[3][v + hl * 8][l16] = acc[v];
    }
    __syncthreads();

    // gate combine: 256 tile elements, 2 per thread
#pragma unroll
    for (int e = tid; e < 256; e += 128) {
        int row = e >> 4, col = e & 15;
        int b = m0 + row, j = j0 + col;
        float rr = sigmoid_f(sAcc[0][row][col] + bi[j]);
        float zz = sigmoid_f(sAcc[1][row][col] + bi[Hsz + j]);
        float nn = tanh_f(sAcc[3][row][col] + bi[2 * Hsz + j] +
                          rr * (sAcc[2][row][col] + bhn[j]));
        float hp = hin[b * Hsz + j];
        float hv = (1.0f - zz) * nn + zz * hp;
        hout[b * Hsz + j] = hv;
        hbf_out[b * Hsz + j] = f2bf(hv);
        out[((long long)b * Ssz + s) * Hsz + j] = hv;
    }
}

// ---------------------------------------------------------------------------
extern "C" void kernel_launch(void* const* d_in, const int* in_sizes, int n_in,
                              void* d_out, int out_size, void* d_ws, size_t ws_size,
                              hipStream_t stream) {
    (void)in_sizes; (void)n_in; (void)out_size; (void)ws_size;

    const long long* tokens = (const long long*)d_in[0];   // [B,S] int64
    const float*     emb    = (const float*)d_in[1];       // [V,D]
    const float*     Wi     = (const float*)d_in[2];       // [D,3H]
    const float*     bi     = (const float*)d_in[3];       // [3H]
    const float*     Wh     = (const float*)d_in[4];       // [H,3H]
    const float*     bhn    = (const float*)d_in[5];       // [H]
    float*           out    = (float*)d_out;               // [B,S,H]

    // workspace layout (byte offsets, all 256-aligned)
    char* ws = (char*)d_ws;
    unsigned short* WiT   = (unsigned short*)(ws);                    //   786432 B
    unsigned short* WhT   = (unsigned short*)(ws + 786432);           //  1572864 B
    float*          hbufs = (float*)(ws + 2359296);                   //   262144 B
    unsigned short* hbf   = (unsigned short*)(ws + 2621440);          //   131072 B
    unsigned short* embbf = (unsigned short*)(ws + 2752512);          // 25600000 B
                                                                      // total ~28.4 MB
    prep_kernel<<<(PREP_TOTAL + 255) / 256, 256, 0, stream>>>(
        Wi, Wh, emb, WiT, WhT, hbufs, hbf, embbf);

    // sequential recurrence: one launch per timestep, double-buffered h
    for (int s = 0; s < Ssz; ++s) {
        const float*    hf_in  = hbufs + (s & 1) * (Bsz * Hsz);
        float*          hf_out = hbufs + ((s + 1) & 1) * (Bsz * Hsz);
        unsigned short* hb_in  = hbf + (s & 1) * (Bsz * Hsz);
        unsigned short* hb_out = hbf + ((s + 1) & 1) * (Bsz * Hsz);
        gru_step_kernel<<<dim3(G3 / 48, Bsz / 16), 128, 0, stream>>>(
            tokens, embbf, WiT, WhT, bi, bhn, hf_in, hb_in, hf_out, hb_out, out, s);
    }
}
